// ReasonNet_8108898255116
// MI455X (gfx1250) — matrix operand order repaired and checked by reading, not compiled, not run on hardware
//
#include <hip/hip_runtime.h>
#include <math.h>

// ---------------------------------------------------------------------------
// ReasonNet block for MI455X (gfx1250, wave32, WMMA).
// GEMMs run on v_wmma_f32_16x16x32_f16 with f32->f16 staging, software
// pipelined: register prefetch of next K-tile + double-buffered LDS.
// sched_barrier pins global-load issue ahead of the WMMA block so HBM
// latency hides under the matrix ops.
// ---------------------------------------------------------------------------

typedef __attribute__((ext_vector_type(16))) _Float16 v16h;
typedef __attribute__((ext_vector_type(8)))  _Float16 v8h;
typedef __attribute__((ext_vector_type(8)))  float    v8f;

#define TILE_M 128
#define TILE_N 128
#define TILE_K 32
#define LDK    40   // padded LDS row stride (halves); multiple of 8 keeps b128 alignment

// ---------------------------------------------------------------------------
// Tiled GEMM: C[M,N] = A[M,Kd] @ B[Kd,N] (+bias) (optional ReLU), f32 in/out,
// f16 WMMA inner product with f32 accumulation.
// Block: 256 threads = 8 waves. Wave grid 4(M) x 2(N); each wave computes
// a 32x64 patch = 2x4 tiles of 16x16.
// Pipeline: tile k+1 global->regs issued before tile k LDS->WMMA compute
// (enforced with sched_barrier); regs->LDS(buf^1) convert/store afterwards;
// one barrier per K-step.
// ---------------------------------------------------------------------------
__global__ __launch_bounds__(256)
void gemm_wmma_f16(const float* __restrict__ A, const float* __restrict__ B,
                   const float* __restrict__ bias, float* __restrict__ C,
                   int M, int N, int Kd, int relu)
{
  __shared__ _Float16 As[2][TILE_M * LDK];
  __shared__ _Float16 Bs[2][TILE_N * LDK];   // stored transposed: Bs[buf][n*LDK + k]

  const int tid  = threadIdx.x;
  const int lane = tid & 31;
  const int wave = tid >> 5;
  const int half = lane >> 4;   // 0: lanes 0-15, 1: lanes 16-31
  const int l16  = lane & 15;
  const int wm   = (wave & 3) * 32;   // wave M offset within block tile
  const int wn   = (wave >> 2) * 64;  // wave N offset within block tile
  const int bm   = blockIdx.y * TILE_M;
  const int bn   = blockIdx.x * TILE_N;

  v8f acc[2][4];
#pragma unroll
  for (int i = 0; i < 2; ++i)
#pragma unroll
    for (int j = 0; j < 4; ++j)
#pragma unroll
      for (int e = 0; e < 8; ++e) acc[i][j][e] = 0.0f;

  // register staging buffers for one K-tile (A: 128x32, B: 32x128)
  float4 ra[4], rb[4];

  // per-thread staging coordinates (constant across K-steps)
  const int arow = tid >> 3;
  const int acol = (tid & 7) << 2;
  const int bkr  = tid >> 5;
  const int bnc  = (tid & 31) << 2;

  // issue all 8 global loads for tile at k-offset kk (no waits here)
  auto load_tile = [&](int kk) {
#pragma unroll
    for (int jj = 0; jj < 4; ++jj)
      ra[jj] = *(const float4*)(A + (size_t)(bm + arow + jj * 32) * Kd + kk + acol);
#pragma unroll
    for (int jj = 0; jj < 4; ++jj)
      rb[jj] = *(const float4*)(B + (size_t)(kk + bkr + jj * 8) * N + bn + bnc);
  };

  // convert f32->f16 and store staged tile into LDS buffer `buf`
  auto store_tile = [&](int buf) {
#pragma unroll
    for (int jj = 0; jj < 4; ++jj) {
      _Float16* p = &As[buf][(arow + jj * 32) * LDK + acol];
      p[0] = (_Float16)ra[jj].x; p[1] = (_Float16)ra[jj].y;
      p[2] = (_Float16)ra[jj].z; p[3] = (_Float16)ra[jj].w;
    }
#pragma unroll
    for (int jj = 0; jj < 4; ++jj) {
      const int kr = bkr + jj * 8;
      Bs[buf][(bnc + 0) * LDK + kr] = (_Float16)rb[jj].x;
      Bs[buf][(bnc + 1) * LDK + kr] = (_Float16)rb[jj].y;
      Bs[buf][(bnc + 2) * LDK + kr] = (_Float16)rb[jj].z;
      Bs[buf][(bnc + 3) * LDK + kr] = (_Float16)rb[jj].w;
    }
  };

  // fragment loads from LDS buffer `buf` + 8 WMMAs
  auto compute = [&](int buf) {
    // A (16x32 f16): lane<16 -> M=l16, K in {0..7,16..23}; lane>=16 -> {8..15,24..31}
    v16h af[2];
#pragma unroll
    for (int i = 0; i < 2; ++i) {
      const _Float16* base = &As[buf][(wm + i * 16 + l16) * LDK + half * 8];
      v8h lo = *(const v8h*)(base);
      v8h hi = *(const v8h*)(base + 16);
#pragma unroll
      for (int e = 0; e < 8; ++e) { af[i][e] = lo[e]; af[i][8 + e] = hi[e]; }
    }
    // B (32x16 f16): lane<16 -> N=l16, K in {0..7,16..23}; lane>=16 -> {8..15,24..31}
    v16h bf[4];
#pragma unroll
    for (int j = 0; j < 4; ++j) {
      const _Float16* base = &Bs[buf][(wn + j * 16 + l16) * LDK + half * 8];
      v8h lo = *(const v8h*)(base);
      v8h hi = *(const v8h*)(base + 16);
#pragma unroll
      for (int e = 0; e < 8; ++e) { bf[j][e] = lo[e]; bf[j][8 + e] = hi[e]; }
    }
#pragma unroll
    for (int i = 0; i < 2; ++i)
#pragma unroll
      for (int j = 0; j < 4; ++j)
        acc[i][j] = __builtin_amdgcn_wmma_f32_16x16x32_f16(
            false, af[i], false, bf[j], (short)0, acc[i][j], false, false);
  };

  // -------- pipeline --------
  load_tile(0);
  store_tile(0);
  __syncthreads();

  int p = 0;
  for (int kk = TILE_K; kk < Kd; kk += TILE_K) {
    load_tile(kk);                       // global loads for next tile in flight...
    __builtin_amdgcn_sched_barrier(0);   // ...pinned ahead of the WMMA block
    compute(p);                          // WMMAs consume current LDS buffer
    store_tile(p ^ 1);                   // loadcnt wait lands here, after compute issued
    __syncthreads();
    p ^= 1;
  }
  compute(p);

  // ---- epilogue: C/D layout (VGPR r: M = half*8 + r, N = l16) ----
#pragma unroll
  for (int i = 0; i < 2; ++i) {
#pragma unroll
    for (int j = 0; j < 4; ++j) {
      int gm0 = bm + wm + i * 16 + half * 8;
      int gn  = bn + wn + j * 16 + l16;
      float bv = bias ? bias[gn] : 0.0f;
#pragma unroll
      for (int r = 0; r < 8; ++r) {
        float val = acc[i][j][r] + bv;
        if (relu) val = fmaxf(val, 0.0f);
        C[(size_t)(gm0 + r) * N + gn] = val;
      }
    }
  }
}

// ---------------------------------------------------------------------------
// Top-K (K=8) neighbor attention. One block (256 threads) per token.
// scores = q . k[nbr] * C^-0.5 + radj ; softmax over 8 ; out = sum w*v[nbr]
// ---------------------------------------------------------------------------
__global__ __launch_bounds__(256)
void attn_topk(const float* __restrict__ q, const float* __restrict__ kf,
               const float* __restrict__ vf, const float* __restrict__ radj,
               const int* __restrict__ inxs, float* __restrict__ out)
{
  const int bt  = blockIdx.x;          // 0 .. B*T-1
  const int b   = bt >> 10;            // T = 1024
  const int tid = threadIdx.x;

  __shared__ float red[8][8];          // [wave][k]
  __shared__ int   nb[8];
  if (tid < 8) nb[tid] = inxs[bt * 8 + tid];
  __syncthreads();

  size_t nbase[8];
#pragma unroll
  for (int k = 0; k < 8; ++k)
    nbase[k] = ((size_t)(b << 10) + nb[k]) * 1024;   // (b*T + idx) * C

  float qv[4];
#pragma unroll
  for (int j = 0; j < 4; ++j) qv[j] = q[(size_t)bt * 1024 + tid + j * 256];

  float p[8];
#pragma unroll
  for (int k = 0; k < 8; ++k) {
    float s = 0.0f;
#pragma unroll
    for (int j = 0; j < 4; ++j) s += qv[j] * kf[nbase[k] + tid + j * 256];
    p[k] = s;
  }
  // wave32 reduction
  for (int off = 16; off; off >>= 1)
#pragma unroll
    for (int k = 0; k < 8; ++k) p[k] += __shfl_xor(p[k], off, 32);
  if ((tid & 31) == 0)
#pragma unroll
    for (int k = 0; k < 8; ++k) red[tid >> 5][k] = p[k];
  __syncthreads();

  float sc[8];
#pragma unroll
  for (int k = 0; k < 8; ++k) {
    float s = 0.0f;
#pragma unroll
    for (int w = 0; w < 8; ++w) s += red[w][k];
    sc[k] = s * 0.03125f + radj[bt * 8 + k];   // 1/sqrt(1024)
  }
  float mx = sc[0];
#pragma unroll
  for (int k = 1; k < 8; ++k) mx = fmaxf(mx, sc[k]);
  float den = 0.0f, wgt[8];
#pragma unroll
  for (int k = 0; k < 8; ++k) { wgt[k] = __expf(sc[k] - mx); den += wgt[k]; }
  float inv = 1.0f / den;
#pragma unroll
  for (int k = 0; k < 8; ++k) wgt[k] *= inv;

#pragma unroll
  for (int j = 0; j < 4; ++j) {
    float o = 0.0f;
#pragma unroll
    for (int k = 0; k < 8; ++k) o += wgt[k] * vf[nbase[k] + tid + j * 256];
    out[(size_t)bt * 1024 + tid + j * 256] = o;
  }
}

// ---------------------------------------------------------------------------
// out = LayerNorm(a + (relu_h ? relu(h) : h)) * g + beta ; one block per row
// ---------------------------------------------------------------------------
__global__ __launch_bounds__(256)
void add_layernorm(const float* __restrict__ a, const float* __restrict__ h,
                   const float* __restrict__ g, const float* __restrict__ beta,
                   float* __restrict__ out, int relu_h)
{
  const int row = blockIdx.x;
  const int tid = threadIdx.x;
  __shared__ float rs[8], rs2[8];

  float v[4];
  float s = 0.0f, s2 = 0.0f;
#pragma unroll
  for (int j = 0; j < 4; ++j) {
    int c = tid + j * 256;
    float hv = h[(size_t)row * 1024 + c];
    if (relu_h) hv = fmaxf(hv, 0.0f);
    float t = a[(size_t)row * 1024 + c] + hv;
    v[j] = t; s += t; s2 += t * t;
  }
  for (int off = 16; off; off >>= 1) {
    s  += __shfl_xor(s,  off, 32);
    s2 += __shfl_xor(s2, off, 32);
  }
  if ((tid & 31) == 0) { rs[tid >> 5] = s; rs2[tid >> 5] = s2; }
  __syncthreads();
  s = 0.0f; s2 = 0.0f;
#pragma unroll
  for (int w = 0; w < 8; ++w) { s += rs[w]; s2 += rs2[w]; }
  const float mu  = s * (1.0f / 1024.0f);
  const float var = s2 * (1.0f / 1024.0f) - mu * mu;
  const float r   = rsqrtf(var + 1e-5f);
#pragma unroll
  for (int j = 0; j < 4; ++j) {
    int c = tid + j * 256;
    out[(size_t)row * 1024 + c] = (v[j] - mu) * r * g[c] + beta[c];
  }
}

// ---------------------------------------------------------------------------
// Launch: 2 hops of (3 GEMMs + attn + GEMM), LN, FFN (2 fused GEMMs), LN.
// Workspace layout (floats), SZ = 16384*1024:
//   [0]   q     (also reused as y)
//   [SZ]  kf    \  reused contiguously as FFN hidden (needs 1.5*SZ)
//   [2SZ] vf    /
//   [3SZ] t     (attn out; also reused as f)
//   [4SZ] h
// total 5*SZ*4 = 320 MB of d_ws.
// ---------------------------------------------------------------------------
extern "C" void kernel_launch(void* const* d_in, const int* in_sizes, int n_in,
                              void* d_out, int out_size, void* d_ws, size_t ws_size,
                              hipStream_t stream) {
  (void)in_sizes; (void)n_in; (void)out_size; (void)ws_size;

  const float* x     = (const float*)d_in[0];
  const float* radj  = (const float*)d_in[1];
  const int*   inxs  = (const int*)  d_in[2];
  const float* Wq    = (const float*)d_in[3];
  const float* Wk    = (const float*)d_in[4];
  const float* Wv    = (const float*)d_in[5];
  const float* Wo    = (const float*)d_in[6];
  const float* ln1_g = (const float*)d_in[7];
  const float* ln1_b = (const float*)d_in[8];
  const float* W1    = (const float*)d_in[9];
  const float* b1    = (const float*)d_in[10];
  const float* W2    = (const float*)d_in[11];
  const float* b2    = (const float*)d_in[12];
  const float* ln2_g = (const float*)d_in[13];
  const float* ln2_b = (const float*)d_in[14];
  float* outp = (float*)d_out;

  const int Mrows = 16 * 1024;       // B*T
  const size_t SZ = (size_t)Mrows * 1024;
  float* ws  = (float*)d_ws;
  float* q   = ws;
  float* kf  = ws + SZ;
  float* vf  = ws + 2 * SZ;
  float* t   = ws + 3 * SZ;
  float* h   = ws + 4 * SZ;
  float* hid = kf;                   // 2*SZ region >= M*1536
  float* y   = q;
  float* f   = t;

  const dim3 blk(256);
  const dim3 gC(1024 / TILE_N, Mrows / TILE_M);   // N=1024 GEMMs
  const dim3 gH(1536 / TILE_N, Mrows / TILE_M);   // N=1536 GEMM

  // ---- hop 1 (input x) ----
  gemm_wmma_f16<<<gC, blk, 0, stream>>>(x, Wq, nullptr, q,  Mrows, 1024, 1024, 0);
  gemm_wmma_f16<<<gC, blk, 0, stream>>>(x, Wk, nullptr, kf, Mrows, 1024, 1024, 0);
  gemm_wmma_f16<<<gC, blk, 0, stream>>>(x, Wv, nullptr, vf, Mrows, 1024, 1024, 0);
  attn_topk<<<Mrows, blk, 0, stream>>>(q, kf, vf, radj, inxs, t);
  gemm_wmma_f16<<<gC, blk, 0, stream>>>(t, Wo, nullptr, h,  Mrows, 1024, 1024, 0);

  // ---- hop 2 (input h) ----
  gemm_wmma_f16<<<gC, blk, 0, stream>>>(h, Wq, nullptr, q,  Mrows, 1024, 1024, 0);
  gemm_wmma_f16<<<gC, blk, 0, stream>>>(h, Wk, nullptr, kf, Mrows, 1024, 1024, 0);
  gemm_wmma_f16<<<gC, blk, 0, stream>>>(h, Wv, nullptr, vf, Mrows, 1024, 1024, 0);
  attn_topk<<<Mrows, blk, 0, stream>>>(q, kf, vf, radj, inxs, t);
  gemm_wmma_f16<<<gC, blk, 0, stream>>>(t, Wo, nullptr, h,  Mrows, 1024, 1024, 0);

  // ---- y = LN(x + relu(h)) ----
  add_layernorm<<<Mrows, blk, 0, stream>>>(x, h, ln1_g, ln1_b, y, 1);

  // ---- FFN: hid = relu(y@W1 + b1) ; f = hid@W2 + b2 ----
  gemm_wmma_f16<<<gH, blk, 0, stream>>>(y,   W1, b1, hid, Mrows, 1536, 1024, 1);
  gemm_wmma_f16<<<gC, blk, 0, stream>>>(hid, W2, b2, f,   Mrows, 1024, 1536, 0);

  // ---- out = LN(y + f) ----
  add_layernorm<<<Mrows, blk, 0, stream>>>(y, f, ln2_g, ln2_b, outp, 0);
}
